// UnconditionalAttentionLayer_24842090840249
// MI455X (gfx1250) — matrix operand order repaired and checked by reading, not compiled, not run on hardware
//
#include <hip/hip_runtime.h>
#include <stdint.h>

// GAT-style multi-head attention, fused flash-attention style, CDNA5.
// N=4096, F_in=512, F_out=128, H=8.
// Wh is stored PRE-SWIZZLED in the exact WMMA B-register layout so the big
// attention GEMM stages it with contiguous async global->LDS b128 copies.

#define N_NODES 4096
#define F_IN    512
#define F_OUT   128
#define N_HEADS 8
#define LEAKK   0.2f
#define NEGINF  (-1e9f)
#define NJBLK   (N_NODES / 32)        // 128 swizzled 32-row blocks per head
#define BLKDW   2048                  // dwords per swizzled 32-row block (8KB)
#define NSTG    (N_NODES / 64)        // 64 stages of 64 K-rows (2 blocks)
#define STGDW   (2 * BLKDW)           // dwords per stage (16KB)

typedef __attribute__((ext_vector_type(16))) __bf16   v16bf;
typedef __attribute__((ext_vector_type(8)))  float    v8f;
typedef __attribute__((ext_vector_type(8)))  uint32_t v8u;
typedef __attribute__((ext_vector_type(4)))  int      v4i;

typedef __attribute__((address_space(1))) v4i gv4i;   // global
typedef __attribute__((address_space(3))) v4i lv4i;   // LDS

#if defined(__has_builtin)
#  if __has_builtin(__builtin_amdgcn_global_load_async_to_lds_b128)
#    define HAVE_ASYNC_LDS 1
#  endif
#  if __has_builtin(__builtin_amdgcn_s_wait_asynccnt)
#    define WAIT_ASYNC(n) __builtin_amdgcn_s_wait_asynccnt(n)
#  endif
#endif
#ifndef WAIT_ASYNC
#  define WAIT_ASYNC(n) asm volatile("s_wait_asynccnt %0" ::"i"(n) : "memory")
#endif

static __device__ __forceinline__ uint32_t pack_bf16(float lo, float hi) {
    __bf16 l = (__bf16)lo, h = (__bf16)hi;
    uint16_t lb = __builtin_bit_cast(uint16_t, l);
    uint16_t hb = __builtin_bit_cast(uint16_t, h);
    return ((uint32_t)hb << 16) | (uint32_t)lb;
}
static __device__ __forceinline__ float bf16_bits_to_f(uint16_t b) {
    return (float)__builtin_bit_cast(__bf16, b);
}

// ---------------------------------------------------------------------------
// Kernel 4 (placed first for disasm visibility): out = softmax(P) . Wh.
// Swizzled Wh stage (64 K-rows, 16KB) is a contiguous async global->LDS copy,
// double-buffered against 16 WMMAs + the A-tile (softmax prob) build.
// ---------------------------------------------------------------------------
__global__ void __launch_bounds__(128)
gat_attn_kernel(const int* __restrict__ adj, const uint32_t* __restrict__ Whz,
                const float* __restrict__ fsrc, const float* __restrict__ fdst,
                const float* __restrict__ mrow, const float* __restrict__ lrow,
                float* __restrict__ out)
{
    __shared__ uint32_t ldsB[2 * STGDW];   // double buffer, 2 x 16KB

    const int tid  = threadIdx.x;
    const int wave = tid >> 5, lane = tid & 31;
    const int h    = blockIdx.y;
    const int i0   = blockIdx.x * 64;
    const int mloc = lane & 15;
    const int row  = i0 + wave * 16 + mloc;
    const int kbaseA = (lane < 16) ? 0 : 8;

    const float fs   = fsrc[h * N_NODES + row];
    const float mr   = mrow[h * N_NODES + row];
    const float invl = 1.0f / lrow[h * N_NODES + row];

    const int*      arow = adj + (size_t)row * N_NODES;
    const float*    fd   = fdst + (size_t)h * N_NODES;
    const uint32_t* gswz = Whz + (size_t)h * NJBLK * BLKDW;

    v8f acc[8] = {};

    // Issue one stage's copy: 8 x b128 per thread, contiguous 16KB.
    auto issue_stage = [&](int sb, int bufsel) {
        const uint32_t* g = gswz + (size_t)sb * STGDW;
        #pragma unroll
        for (int q = 0; q < 8; ++q) {
            int off = q * 512 + tid * 4;               // dwords
#ifdef HAVE_ASYNC_LDS
            __builtin_amdgcn_global_load_async_to_lds_b128(
                (gv4i*)(g + off),
                (lv4i*)(&ldsB[bufsel * STGDW + off]),
                0, 0);
#else
            uint4 d = *reinterpret_cast<const uint4*>(g + off);
            *reinterpret_cast<uint4*>(&ldsB[bufsel * STGDW + off]) = d;
#endif
        }
    };

    issue_stage(0, 0);   // prologue

    for (int sb = 0; sb < NSTG; ++sb) {
        __syncthreads();              // all waves done reading buf[(sb+1)&1]
        if (sb + 1 < NSTG) {
            issue_stage(sb + 1, (sb + 1) & 1);
#ifdef HAVE_ASYNC_LDS
            WAIT_ASYNC(8);            // stage sb's 8 transfers complete
#endif
        } else {
#ifdef HAVE_ASYNC_LDS
            WAIT_ASYNC(0);
#endif
        }
        __syncthreads();              // buf[sb&1] visible to all waves

        const uint32_t* lb = &ldsB[(sb & 1) * STGDW];

        #pragma unroll
        for (int sub = 0; sub < 2; ++sub) {
            const int j = sb * 64 + sub * 32;

            // Normalized attention-prob tile P (16x32) in A layout.
            v8u au;
            #pragma unroll
            for (int v = 0; v < 8; ++v) {
                int k = 2 * kbaseA + 2 * v;
                int2   aa = *reinterpret_cast<const int2*>(arow + j + k);
                float2 ff = *reinterpret_cast<const float2*>(fd + j + k);
                float t0 = fs + ff.x; t0 = (t0 > 0.f) ? t0 : LEAKK * t0;
                float t1 = fs + ff.y; t1 = (t1 > 0.f) ? t1 : LEAKK * t1;
                float p0 = (aa.x > 0) ? __expf(t0 - mr) * invl : 0.f;
                float p1 = (aa.y > 0) ? __expf(t1 - mr) * invl : 0.f;
                au[v] = pack_bf16(p0, p1);
            }
            v16bf A = __builtin_bit_cast(v16bf, au);

            const uint32_t* lsub = lb + sub * BLKDW;
            v16bf Bt[8];
            #pragma unroll
            for (int nt = 0; nt < 8; ++nt)
                Bt[nt] = *reinterpret_cast<const v16bf*>(&lsub[(nt * 32 + lane) * 8]);
            #pragma unroll
            for (int nt = 0; nt < 8; ++nt)
                acc[nt] = __builtin_amdgcn_wmma_f32_16x16x32_bf16(
                    false, A, false, Bt[nt], (short)0, acc[nt], false, false);
        }
    }

    const int rOff = (lane >= 16) ? 8 : 0;
    const int nn = lane & 15;
    #pragma unroll
    for (int nt = 0; nt < 8; ++nt) {
        #pragma unroll
        for (int r = 0; r < 8; ++r) {
            int i = i0 + wave * 16 + rOff + r;
            out[(size_t)i * (N_HEADS * F_OUT) + h * F_OUT + nt * 16 + nn] = acc[nt][r];
        }
    }
}

// ---------------------------------------------------------------------------
// Kernel 1: Wh = x . W[h]  (bf16 WMMA, fp32 acc), output written pre-swizzled
// into B-register layout: dword[(((h*128+jblk)*8+nt)*32+L)*8+v] packs node
// rows (jblk*32 + kb + 2v, +1) at column nt*16 + (L&15), kb = (L<16?0:16).
// ---------------------------------------------------------------------------
__global__ void __launch_bounds__(128)
gat_wh_kernel(const float* __restrict__ x, const float* __restrict__ W,
              uint32_t* __restrict__ Whz)
{
    __shared__ uint32_t ldsB[2048];

    const int tid  = threadIdx.x;
    const int wave = tid >> 5, lane = tid & 31;
    const int h    = blockIdx.y;
    const int i0   = blockIdx.x * 64;
    const int mloc = lane & 15;
    const int row  = i0 + wave * 16 + mloc;
    const int kbaseA = (lane < 16) ? 0 : 8;

    v8f acc[8] = {};

    for (int ks = 0; ks < F_IN; ks += 32) {
        __syncthreads();
        // Stage B = W[h][ks..ks+32)[0..128) into B-register layout in LDS.
        {
            const int col = tid;
            const int nt = col >> 4, nn = col & 15;
            const float* src = W + ((size_t)h * F_IN + ks) * F_OUT + col;
            #pragma unroll
            for (int kp = 0; kp < 16; ++kp) {
                float b0 = src[(2 * kp) * F_OUT];
                float b1 = src[(2 * kp + 1) * F_OUT];
                int L = nn + ((kp >= 8) ? 16 : 0);
                ldsB[(nt * 32 + L) * 8 + (kp & 7)] = pack_bf16(b0, b1);
            }
        }
        __syncthreads();
        // A tile (16x32 of x) in A-register layout.
        v8u au;
        const float* xrow = x + (size_t)row * F_IN + ks;
        #pragma unroll
        for (int v = 0; v < 8; ++v) {
            int k = 2 * kbaseA + 2 * v;
            float2 xx = *reinterpret_cast<const float2*>(xrow + k);
            au[v] = pack_bf16(xx.x, xx.y);
        }
        v16bf A = __builtin_bit_cast(v16bf, au);
        v16bf Bt[8];
        #pragma unroll
        for (int nt = 0; nt < 8; ++nt)
            Bt[nt] = *reinterpret_cast<const v16bf*>(&ldsB[(nt * 32 + lane) * 8]);
        #pragma unroll
        for (int nt = 0; nt < 8; ++nt)
            acc[nt] = __builtin_amdgcn_wmma_f32_16x16x32_bf16(
                false, A, false, Bt[nt], (short)0, acc[nt], false, false);
    }

    // Write swizzled output. Lane owns 8 consecutive node rows starting at
    // base_row (multiple of 8) -> same jblk, same K-half, 4 consecutive v's.
    const int rOff = (lane >= 16) ? 8 : 0;
    const int nn = lane & 15;
    const int base_row = i0 + wave * 16 + rOff;
    const int jblk = base_row >> 5;
    const int kin  = base_row & 31;
    const int Lw   = nn + ((kin & 16) ? 16 : 0);
    const int vb   = (kin & 15) >> 1;
    #pragma unroll
    for (int nt = 0; nt < 8; ++nt) {
        uint4 d;
        d.x = pack_bf16(acc[nt][0], acc[nt][1]);
        d.y = pack_bf16(acc[nt][2], acc[nt][3]);
        d.z = pack_bf16(acc[nt][4], acc[nt][5]);
        d.w = pack_bf16(acc[nt][6], acc[nt][7]);
        uint32_t* dst = Whz + ((((size_t)h * NJBLK + jblk) * 8 + nt) * 32 + Lw) * 8 + vb;
        *reinterpret_cast<uint4*>(dst) = d;   // 16B aligned (vb is 0 or 4)
    }
}

// ---------------------------------------------------------------------------
// Kernel 2: f_src/f_dst dot products, reading the swizzled Wh layout. Tiny.
// ---------------------------------------------------------------------------
__global__ void __launch_bounds__(256)
gat_feat_kernel(const uint32_t* __restrict__ Whz,
                const float* __restrict__ a_src, const float* __restrict__ a_dst,
                float* __restrict__ fsrc, float* __restrict__ fdst)
{
    int idx = blockIdx.x * 256 + threadIdx.x;      // h*N + i
    if (idx >= N_HEADS * N_NODES) return;
    int h = idx >> 12;
    int i = idx & (N_NODES - 1);
    int jblk = i >> 5, kin = i & 31;
    int Lh = (kin & 16) ? 16 : 0;
    int v  = (kin & 15) >> 1;
    int hi = kin & 1;
    const uint32_t* basep = Whz + ((size_t)h * NJBLK + jblk) * BLKDW;
    const float* as = a_src + h * F_OUT;
    const float* ad = a_dst + h * F_OUT;
    float s0 = 0.f, s1 = 0.f;
    for (int nt = 0; nt < 8; ++nt) {
        #pragma unroll
        for (int cc = 0; cc < 16; ++cc) {
            uint32_t d = basep[(size_t)(nt * 32 + cc + Lh) * 8 + v];
            float w = bf16_bits_to_f((uint16_t)(hi ? (d >> 16) : (d & 0xffff)));
            int c = nt * 16 + cc;
            s0 += w * as[c];
            s1 += w * ad[c];
        }
    }
    fsrc[idx] = s0;
    fdst[idx] = s1;
}

// ---------------------------------------------------------------------------
// Kernel 3: per (h,row) softmax stats. One wave per row, coalesced adj reads.
// ---------------------------------------------------------------------------
__global__ void __launch_bounds__(128)
gat_stats_kernel(const int* __restrict__ adj,
                 const float* __restrict__ fsrc, const float* __restrict__ fdst,
                 float* __restrict__ mrow, float* __restrict__ lrow)
{
    int g    = blockIdx.x * 4 + (threadIdx.x >> 5);  // h*N + i
    int lane = threadIdx.x & 31;
    int h = g >> 12;
    int i = g & (N_NODES - 1);
    float fs = fsrc[g];
    const int*   arow = adj + (size_t)i * N_NODES;
    const float* fd   = fdst + (size_t)h * N_NODES;

    float m = -1e30f, s = 0.f;
    for (int j = lane; j < N_NODES; j += 32) {
        float e;
        if (arow[j] > 0) {
            float t = fs + fd[j];
            e = (t > 0.f) ? t : LEAKK * t;
        } else {
            e = NEGINF;
        }
        float mn = fmaxf(m, e);
        s = s * __expf(m - mn) + __expf(e - mn);
        m = mn;
    }
    #pragma unroll
    for (int off = 16; off > 0; off >>= 1) {
        float mo = __shfl_xor(m, off, 32);
        float so = __shfl_xor(s, off, 32);
        float mn = fmaxf(m, mo);
        s = s * __expf(m - mn) + so * __expf(mo - mn);
        m = mn;
    }
    if (lane == 0) { mrow[g] = m; lrow[g] = s; }
}

// ---------------------------------------------------------------------------
extern "C" void kernel_launch(void* const* d_in, const int* in_sizes, int n_in,
                              void* d_out, int out_size, void* d_ws, size_t ws_size,
                              hipStream_t stream)
{
    const float* x     = (const float*)d_in[0];   // [4096,512]
    const int*   adj   = (const int*)  d_in[1];   // [4096,4096]
    const float* W     = (const float*)d_in[2];   // [8,512,128]
    const float* a_src = (const float*)d_in[3];   // [8,128]
    const float* a_dst = (const float*)d_in[4];   // [8,128]
    float* out = (float*)d_out;                   // [4096,1024]

    char* ws = (char*)d_ws;
    uint32_t* Whz = (uint32_t*)ws;                           // 8 MiB swizzled Wh
    float* fsrc = (float*)(ws + (size_t)N_HEADS * NJBLK * BLKDW * 4);
    float* fdst = fsrc + N_HEADS * N_NODES;
    float* mr   = fdst + N_HEADS * N_NODES;
    float* lr   = mr   + N_HEADS * N_NODES;

    dim3 blk(128);
    dim3 gTile(N_NODES / 64, N_HEADS);

    gat_wh_kernel<<<gTile, blk, 0, stream>>>(x, W, Whz);
    gat_feat_kernel<<<(N_HEADS * N_NODES) / 256, 256, 0, stream>>>(Whz, a_src, a_dst, fsrc, fdst);
    gat_stats_kernel<<<(N_HEADS * N_NODES) / 4, 128, 0, stream>>>(adj, fsrc, fdst, mr, lr);
    gat_attn_kernel<<<gTile, blk, 0, stream>>>(adj, Whz, fsrc, fdst, mr, lr, out);
}
